// TitleGenerator_35519379538042
// MI455X (gfx1250) — matrix operand order repaired and checked by reading, not compiled, not run on hardware
//
#include <hip/hip_runtime.h>

typedef __attribute__((ext_vector_type(16))) __bf16 v16bf;
typedef __attribute__((ext_vector_type(8)))  __bf16 v8bf;
typedef __attribute__((ext_vector_type(8)))  float  v8f;

#define B_    2048
#define D_    4096
#define H_    128
#define C_    96
#define T_    400
#define THR_  (1.0f - 1e-5f)

// ---- workspace byte offsets ----
#define LIN1F_OFF 0ull                          // 8 ntiles * 128 ksteps * 512 elems * 2B = 1 MiB
#define WIHF_OFF  1048576ull                    // 24*4*512*2 = 96 KiB
#define WHHF_OFF  (WIHF_OFF + 98304ull)         // 96 KiB
#define WL2F_OFF  (WHHF_OFF + 98304ull)         // 6*4*512*2 = 24 KiB
#define X0_OFF    (WL2F_OFF + 24576ull)         // 2048*128*2 = 512 KiB

// ---- LDS byte offsets (recurrence kernel) ----
#define SM_WIH   0
#define SM_WHH   98304
#define SM_WL2   196608
#define SM_BIH   221184
#define SM_BHH   222720
#define SM_L2B   224256
#define SM_WAVE  224640
#define WAVE_BYTES 13056          // Xb0(4352) + Xb1(4352) + Lb(4352)
#define XROW     136              // padded row (128 + 8 bf16) to spread LDS banks
#define SMEM_TOTAL (SM_WAVE + 4*WAVE_BYTES)   // 276864 B < 320 KiB

static __device__ __forceinline__ v8f wmma_bf16(v16bf a, v16bf b, v8f c) {
  return __builtin_amdgcn_wmma_f32_16x16x32_bf16(false, a, false, b, (short)0, c, false, false);
}
static __device__ __forceinline__ float sigm(float x) { return 1.0f / (1.0f + expf(-x)); }
static __device__ __forceinline__ float leaky(float x) { return x >= 0.0f ? x : 0.01f * x; }

// Async global->LDS 16B copy (CDNA5 GLOBAL_LOAD_ASYNC_TO_LDS_B128, GV mode, ASYNCcnt).
// ldsOff: byte offset in LDS (low 32 bits of the generic shared pointer).
static __device__ __forceinline__ void async_g2l_b128(unsigned ldsOff, unsigned long long gaddr) {
  asm volatile("global_load_async_to_lds_b128 %0, %1, off" :: "v"(ldsOff), "v"(gaddr) : "memory");
}
static __device__ __forceinline__ void wait_async0() {
  asm volatile("s_wait_asynccnt 0" ::: "memory");
}

#define SHUF16(lo, hi) __builtin_shufflevector(lo, hi, 0,1,2,3,4,5,6,7,8,9,10,11,12,13,14,15)

// Swizzle a row-major f32 weight W[ntiles*16][ldw] into per-lane bf16 WMMA B-fragments:
// frag(ntile,kstep): lane l holds N=l&15, K=(l<16?0:8)+(e&7)+(e<8?0:16) for e=0..15, 32B contiguous per lane.
__global__ void __launch_bounds__(256) make_bfrags(const float* __restrict__ W,
                                                   __bf16* __restrict__ dst,
                                                   int ntiles, int ksteps, int ldw) {
  int idx = blockIdx.x * 256 + threadIdx.x;
  int total = ntiles * ksteps * 512;
  if (idx >= total) return;
  int e    = idx & 15;
  int lane = (idx >> 4) & 31;
  int frag = idx >> 9;
  int kstep = frag % ksteps;
  int ntile = frag / ksteps;
  int n = lane & 15;
  int k = kstep * 32 + ((lane >> 4) << 3) + (e & 7) + ((e >> 3) << 4);
  dst[idx] = (__bf16)W[(size_t)(ntile * 16 + n) * ldw + k];
}

// Phase 1: X0 = leaky(img_feat @ lin1_w.T + lin1_b), bf16 out. One wave per 16x16 tile.
__global__ void __launch_bounds__(128) phase1_kernel(const float* __restrict__ img,
                                                     const __bf16* __restrict__ lin1F,
                                                     const float* __restrict__ lin1_b,
                                                     __bf16* __restrict__ X0) {
  int wave = (blockIdx.x * blockDim.x + threadIdx.x) >> 5;   // 0..1023
  int lane = threadIdx.x & 31;
  int mtile = wave >> 3, ntile = wave & 7;
  int lh = lane >> 4, n = lane & 15;
  const float* arow = img + (size_t)(mtile * 16 + n) * D_;
  v8f acc = {};
  for (int ks = 0; ks < 128; ++ks) {
    int kb = ks * 32 + (lh ? 8 : 0);
    const float4* p0 = (const float4*)(arow + kb);
    const float4* p1 = (const float4*)(arow + kb + 16);
    float4 a0 = p0[0], a1 = p0[1], b0 = p1[0], b1 = p1[1];
    v16bf A = { (__bf16)a0.x, (__bf16)a0.y, (__bf16)a0.z, (__bf16)a0.w,
                (__bf16)a1.x, (__bf16)a1.y, (__bf16)a1.z, (__bf16)a1.w,
                (__bf16)b0.x, (__bf16)b0.y, (__bf16)b0.z, (__bf16)b0.w,
                (__bf16)b1.x, (__bf16)b1.y, (__bf16)b1.z, (__bf16)b1.w };
    v16bf Bf = *(const v16bf*)(lin1F + (((size_t)(ntile * 128 + ks) * 32 + lane) << 4));
    if (ks + 1 < 128)
      __builtin_prefetch(lin1F + (((size_t)(ntile * 128 + ks + 1) * 32 + lane) << 4), 0, 0);
    acc = wmma_bf16(A, Bf, acc);
  }
  int col = ntile * 16 + n;
  float bias = lin1_b[col];
#pragma unroll
  for (int r = 0; r < 8; ++r) {
    int m = mtile * 16 + r + lh * 8;
    X0[(size_t)m * H_ + col] = (__bf16)leaky(acc[r] + bias);
  }
}

// Persistent recurrence: 32 blocks x 4 waves; each wave owns a 16-row batch tile
// for all 400 timesteps. Weights live in LDS (bf16 fragments, staged with
// async global->LDS DMA), activations double-buffered in LDS.
__global__ void __launch_bounds__(128) recurrence_kernel(
    const __bf16* __restrict__ wfrags,   // Wih|Whh|Wl2 fragments, contiguous (221184 B)
    const __bf16* __restrict__ X0,
    const float* __restrict__ b_ih, const float* __restrict__ b_hh,
    const float* __restrict__ l2b,
    float* __restrict__ titles, int* __restrict__ lens) {
  extern __shared__ char smem[];
  __bf16* sWih = (__bf16*)(smem + SM_WIH);
  __bf16* sWhh = (__bf16*)(smem + SM_WHH);
  __bf16* sWl2 = (__bf16*)(smem + SM_WL2);
  float*  sBih = (float*)(smem + SM_BIH);
  float*  sBhh = (float*)(smem + SM_BHH);
  float*  sL2b = (float*)(smem + SM_L2B);

  int tid = threadIdx.x, lane = tid & 31, wv = tid >> 5;
  int lh = lane >> 4, n = lane & 15;
  int bt = blockIdx.x * 4 + wv;                 // batch tile 0..127

  // LDS byte offsets via flat-aperture low bits of the generic pointer
  unsigned smemOff = (unsigned)(size_t)smem;
  __bf16* Xb0 = (__bf16*)(smem + SM_WAVE + wv * WAVE_BYTES);
  __bf16* Xb1 = Xb0 + 16 * XROW;
  __bf16* Lb  = Xb1 + 16 * XROW;

  { // async DMA weight staging into LDS (no VGPR round-trip, ASYNCcnt-tracked)
    unsigned long long gsrc = (unsigned long long)(size_t)wfrags;
    for (int i = tid; i < 13824; i += 128)                    // 221184 B of fragments
      async_g2l_b128(smemOff + (unsigned)i * 16u, gsrc + (unsigned long long)i * 16ull);
    // per-wave h0 tile: 16 rows x 256 B global -> padded 272 B LDS rows
    unsigned xb0Off = smemOff + SM_WAVE + (unsigned)wv * WAVE_BYTES;
    unsigned long long gX0 = (unsigned long long)(size_t)(X0 + (size_t)bt * 16 * H_);
    for (int i = lane; i < 256; i += 32) {
      int row = i >> 4, c16 = i & 15;
      async_g2l_b128(xb0Off + (unsigned)(row * XROW * 2 + c16 * 16),
                     gX0 + (unsigned long long)(row * 256 + c16 * 16));
    }
    // biases (small, regular loads)
    for (int i = tid; i < 384; i += 128) { sBih[i] = b_ih[i]; sBhh[i] = b_hh[i]; }
    if (tid < 96) sL2b[tid] = l2b[tid];
    wait_async0();
  }
  __syncthreads();

  int lenr[8];
#pragma unroll
  for (int r = 0; r < 8; ++r) lenr[r] = T_;

  for (int t = 0; t < T_; ++t) {
    __bf16* Xc = (t & 1) ? Xb1 : Xb0;
    __bf16* Xn = (t & 1) ? Xb0 : Xb1;

    // A fragments of current state (x == h for t>0; h==0 at t==0)
    v16bf Af[4];
#pragma unroll
    for (int ks = 0; ks < 4; ++ks) {
      int k0 = ks * 32 + (lh ? 8 : 0);
      const __bf16* p = Xc + n * XROW + k0;
      v8bf lo = *(const v8bf*)p, hi = *(const v8bf*)(p + 16);
      Af[ks] = SHUF16(lo, hi);
    }

#pragma unroll 1
    for (int j = 0; j < 8; ++j) {               // hidden-column tiles
      v8f ir = {}, iz = {}, inn = {}, hr = {}, hz = {}, hnn = {};
#pragma unroll
      for (int ks = 0; ks < 4; ++ks) {
        v16bf a = Af[ks];
        ir  = wmma_bf16(a, *(const v16bf*)(sWih + ((((j     ) * 4 + ks) * 32 + lane) << 4)), ir);
        iz  = wmma_bf16(a, *(const v16bf*)(sWih + ((((j +  8) * 4 + ks) * 32 + lane) << 4)), iz);
        inn = wmma_bf16(a, *(const v16bf*)(sWih + ((((j + 16) * 4 + ks) * 32 + lane) << 4)), inn);
        if (t > 0) {
          hr  = wmma_bf16(a, *(const v16bf*)(sWhh + ((((j     ) * 4 + ks) * 32 + lane) << 4)), hr);
          hz  = wmma_bf16(a, *(const v16bf*)(sWhh + ((((j +  8) * 4 + ks) * 32 + lane) << 4)), hz);
          hnn = wmma_bf16(a, *(const v16bf*)(sWhh + ((((j + 16) * 4 + ks) * 32 + lane) << 4)), hnn);
        }
      }
      int col = j * 16 + n;
      float bir = sBih[col],       bhr = sBhh[col];
      float biz = sBih[128 + col], bhz = sBhh[128 + col];
      float bin = sBih[256 + col], bhn = sBhh[256 + col];
#pragma unroll
      for (int r = 0; r < 8; ++r) {
        int grow = r + lh * 8;
        float rv = sigm((ir[r] + bir) + (hr[r] + bhr));
        float zv = sigm((iz[r] + biz) + (hz[r] + bhz));
        float nv = tanhf((inn[r] + bin) + rv * (hnn[r] + bhn));
        float hp = (t == 0) ? 0.0f : (float)Xc[grow * XROW + col];
        float hn = (1.0f - zv) * nv + zv * hp;
        Xn[grow * XROW + col] = (__bf16)hn;
        Lb[grow * XROW + col] = (__bf16)leaky(hn);
      }
    }
    // cross-lane LDS handoff within the wave: drain DS before reloading as A-fragments
    asm volatile("s_wait_dscnt 0" ::: "memory");

    v16bf Lf[4];
#pragma unroll
    for (int ks = 0; ks < 4; ++ks) {
      int k0 = ks * 32 + (lh ? 8 : 0);
      const __bf16* p = Lb + n * XROW + k0;
      v8bf lo = *(const v8bf*)p, hi = *(const v8bf*)(p + 16);
      Lf[ks] = SHUF16(lo, hi);
    }
    v8f cc[6];
#pragma unroll
    for (int c = 0; c < 6; ++c) {
      v8f acc = {};
#pragma unroll
      for (int ks = 0; ks < 4; ++ks)
        acc = wmma_bf16(Lf[ks], *(const v16bf*)(sWl2 + (((c * 4 + ks) * 32 + lane) << 4)), acc);
      float bb = sL2b[c * 16 + n];
#pragma unroll
      for (int r = 0; r < 8; ++r) acc[r] += bb;
      cc[c] = acc;
    }
#pragma unroll
    for (int r = 0; r < 8; ++r) {
      float mv = cc[0][r];
#pragma unroll
      for (int c = 1; c < 6; ++c) mv = fmaxf(mv, cc[c][r]);
#pragma unroll
      for (int off = 8; off >= 1; off >>= 1)
        mv = fmaxf(mv, __shfl_xor(mv, off, 16));   // row max across 16-lane half
      int grow = r + lh * 8;
      float* op = titles + ((size_t)(bt * 16 + grow) * T_ + t) * C_;
#pragma unroll
      for (int c = 0; c < 6; ++c) {
        float v = cc[c][r] / mv;
        v = (v > THR_) ? v : 0.0f;
        op[c * 16 + n] = v;
        if (c == 3 && n == 4) {                   // column 52 == NULL_TERM_IND
          if (v == 1.0f && lenr[r] == T_) lenr[r] = t + 1;
        }
      }
    }
  }
  if (n == 4) {
#pragma unroll
    for (int r = 0; r < 8; ++r) lens[bt * 16 + r + lh * 8] = lenr[r];
  }
}

extern "C" void kernel_launch(void* const* d_in, const int* in_sizes, int n_in,
                              void* d_out, int out_size, void* d_ws, size_t ws_size,
                              hipStream_t stream) {
  (void)in_sizes; (void)n_in; (void)out_size; (void)ws_size;
  const float* img    = (const float*)d_in[0];
  const float* lin1_w = (const float*)d_in[1];
  const float* lin1_b = (const float*)d_in[2];
  const float* w_ih   = (const float*)d_in[3];
  const float* w_hh   = (const float*)d_in[4];
  const float* b_ih   = (const float*)d_in[5];
  const float* b_hh   = (const float*)d_in[6];
  const float* lin2_w = (const float*)d_in[7];
  const float* lin2_b = (const float*)d_in[8];

  char* ws = (char*)d_ws;
  __bf16* lin1F  = (__bf16*)(ws + LIN1F_OFF);
  __bf16* wfrags = (__bf16*)(ws + WIHF_OFF);
  __bf16* WihF   = (__bf16*)(ws + WIHF_OFF);
  __bf16* WhhF   = (__bf16*)(ws + WHHF_OFF);
  __bf16* Wl2F   = (__bf16*)(ws + WL2F_OFF);
  __bf16* X0     = (__bf16*)(ws + X0_OFF);

  float* titles = (float*)d_out;
  int*   lens   = (int*)(titles + (size_t)B_ * T_ * C_);

  make_bfrags<<<(8 * 128 * 512 + 255) / 256, 256, 0, stream>>>(lin1_w, lin1F, 8, 128, 4096);
  make_bfrags<<<(24 * 4 * 512 + 255) / 256, 256, 0, stream>>>(w_ih, WihF, 24, 4, 128);
  make_bfrags<<<(24 * 4 * 512 + 255) / 256, 256, 0, stream>>>(w_hh, WhhF, 24, 4, 128);
  make_bfrags<<<(6 * 4 * 512 + 255) / 256, 256, 0, stream>>>(lin2_w, Wl2F, 6, 4, 128);
  phase1_kernel<<<256, 128, 0, stream>>>(img, lin1F, lin1_b, X0);
  recurrence_kernel<<<32, 128, SMEM_TOTAL, stream>>>(wfrags, X0, b_ih, b_hh, lin2_b,
                                                     titles, lens);
}
